// AttentionBlock_15118284882511
// MI455X (gfx1250) — matrix operand order repaired and checked by reading, not compiled, not run on hardware
//
#include <hip/hip_runtime.h>

// ---------------------------------------------------------------------------
// Causal single-head attention, B=4, T=4096, C=256 (fp32 in/out).
// fp16 WMMA (v_wmma_f32_16x16x32_f16) + flash-style online softmax.
// Block = 4 waves = 64 Q rows; K and V staged via async global->LDS b128,
// V transposed LDS->LDS with packed b32 stores. All LDS strides padded so
// B-operand ds_load_b128 reads are bank-conflict-free.
// ---------------------------------------------------------------------------

typedef __attribute__((ext_vector_type(16))) _Float16 v16h;
typedef __attribute__((ext_vector_type(8)))  _Float16 v8h;
typedef __attribute__((ext_vector_type(8)))  float    v8f;
typedef __attribute__((ext_vector_type(4)))  int      v4i;

typedef __attribute__((address_space(1))) v4i* gv4i_p;   // global int4*
typedef __attribute__((address_space(3))) v4i* lv4i_p;   // LDS int4*

#define WMMA16(a, b, c) \
    __builtin_amdgcn_wmma_f32_16x16x32_f16(false, (a), false, (b), (short)0, (c), false, false)

#if defined(__has_builtin)
#if __has_builtin(__builtin_amdgcn_global_load_async_to_lds_b128)
#define HAVE_ASYNC_LDS 1
#endif
#endif

static constexpr int kB = 4;
static constexpr int kT = 4096;
static constexpr int kC = 256;
static constexpr int kRows = kB * kT;          // 16384
static constexpr int kKsStride = 264;          // halves: 132 dw -> lid*4 banks, 16B rows
static constexpr int kVtStride = 40;           // halves: 20 dw  -> lid*20 banks, 16B rows
static constexpr int kPStride  = 40;           // halves: same trick for P repack

static __device__ __forceinline__ v8f zero8() {
    v8f z;
#pragma unroll
    for (int i = 0; i < 8; ++i) z[i] = 0.0f;
    return z;
}

// 16 contiguous halves (two 16B LDS/global loads)
static __device__ __forceinline__ v16h load16c(const _Float16* p) {
    const v8h lo = *(const v8h*)p;
    const v8h hi = *(const v8h*)(p + 8);
    v16h r;
#pragma unroll
    for (int i = 0; i < 8; ++i) { r[i] = lo[i]; r[i + 8] = hi[i]; }
    return r;
}

// ---------------------------------------------------------------------------
// Kernel 1: q/k/v = x @ W^T + b  ->  f16 [kRows, kC] in workspace.
// ---------------------------------------------------------------------------
__global__ __launch_bounds__(256) void qkv_proj_kernel(
    const float* __restrict__ x,
    const float* __restrict__ Wq, const float* __restrict__ bq,
    const float* __restrict__ Wk, const float* __restrict__ bk,
    const float* __restrict__ Wv, const float* __restrict__ bv,
    _Float16* __restrict__ qh, _Float16* __restrict__ kh, _Float16* __restrict__ vh)
{
    const int lane    = threadIdx.x & 31;
    const int wave    = threadIdx.x >> 5;
    const int mt      = blockIdx.x * 8 + wave;
    const int nt      = blockIdx.y;
    const int t0      = mt * 16;
    const int d0      = nt * 16;
    const int half_id = lane >> 4;
    const int lid     = lane & 15;

    const float* W;  const float* bias;  _Float16* out;
    if      (blockIdx.z == 0) { W = Wq; bias = bq; out = qh; }
    else if (blockIdx.z == 1) { W = Wk; bias = bk; out = kh; }
    else                      { W = Wv; bias = bv; out = vh; }

    v8f acc;
    {
        const float bval = bias[d0 + lid];
#pragma unroll
        for (int r = 0; r < 8; ++r) acc[r] = bval;
    }

    const float* xrow  = x + (size_t)(t0 + lid) * kC;
    const float* wrow  = W + (size_t)(d0 + lid) * kC;
    const int    abase = half_id * 16;
    const int    bbase = half_id * 16;

    for (int kk = 0; kk < kC; kk += 32) {
        v16h a, b;
#pragma unroll
        for (int i = 0; i < 8; ++i) {
            a[i]     = (_Float16)xrow[kk + abase + i];
            a[i + 8] = (_Float16)xrow[kk + abase + 8 + i];
        }
#pragma unroll
        for (int i = 0; i < 16; ++i)
            b[i] = (_Float16)wrow[kk + bbase + i];
        acc = WMMA16(a, b, acc);
    }

#pragma unroll
    for (int r = 0; r < 8; ++r) {
        const int t = t0 + r + half_id * 8;
        out[(size_t)t * kC + d0 + lid] = (_Float16)acc[r];
    }
}

// ---------------------------------------------------------------------------
// Kernel 2: flash attention.
// ---------------------------------------------------------------------------
__global__ __launch_bounds__(128) void flash_attn_kernel(
    const _Float16* __restrict__ qh,
    const _Float16* __restrict__ kh,
    const _Float16* __restrict__ vh,
    float* __restrict__ out)
{
    __shared__ _Float16 ks[32 * kKsStride];        // 16.5KB K tile [s][d] (padded)
    __shared__ _Float16 vrow[32 * kKsStride];      // 16.5KB V tile [s][d] (padded)
    __shared__ _Float16 vt[kC * kVtStride];        // 20KB   V^T   [d][s] (padded)
    __shared__ _Float16 pstage[4][16 * kPStride];  // 5KB    per-wave P repack

    const int tid     = threadIdx.x;               // 0..127
    const int lane    = tid & 31;
    const int wave    = tid >> 5;
    const int row0    = blockIdx.x * 64;
    const int bidx    = row0 >> 12;                // batch (4096 rows each)
    const int t0b     = row0 & (kT - 1);
    const int t0      = t0b + wave * 16;
    const size_t rowbase = (size_t)bidx * kT;
    const int half_id = lane >> 4;
    const int lid     = lane & 15;
    const int abase   = half_id * 8;
    const int bbase   = half_id * 16;

    // ---- preload Q tile in A layout (64 VGPRs) ----
    v16h qa[8];
    {
        const _Float16* qrow = qh + (rowbase + t0 + lid) * kC;
#pragma unroll
        for (int kk = 0; kk < 8; ++kk) {
            const v8h lo = *(const v8h*)(qrow + kk * 32 + bbase);
            const v8h hi = *(const v8h*)(qrow + kk * 32 + bbase + 8);
#pragma unroll
            for (int i = 0; i < 8; ++i) { qa[kk][i] = lo[i]; qa[kk][i + 8] = hi[i]; }
        }
    }

    v8f oacc[16];
#pragma unroll
    for (int n = 0; n < 16; ++n) oacc[n] = zero8();
    float mrow[8], lrow[8];
#pragma unroll
    for (int r = 0; r < 8; ++r) { mrow[r] = -1e30f; lrow[r] = 0.0f; }

    const float scale = 0.0625f;                   // 256^-0.5
    _Float16* pl = pstage[wave];
    const int nsteps = (t0b + 64) >> 5;            // uniform across the block

    for (int js = 0; js < nsteps; ++js) {
        const int s0 = js * 32;

        __syncthreads();   // A: previous step's reads of ks/vrow/vt complete

        // ---- stage K and V tiles (row-major): 1024 x 16B chunks each ----
        {
            const _Float16* kbase = kh + (rowbase + s0) * kC;
            const _Float16* vbase = vh + (rowbase + s0) * kC;
#pragma unroll
            for (int rep = 0; rep < 8; ++rep) {
                const int c  = rep * 128 + tid;
                const int s  = c >> 5;
                const int ch = c & 31;
#if HAVE_ASYNC_LDS
                __builtin_amdgcn_global_load_async_to_lds_b128(
                    (gv4i_p)(v4i*)const_cast<_Float16*>(kbase + s * kC + ch * 8),
                    (lv4i_p)(v4i*)(ks + s * kKsStride + ch * 8), 0, 0);
                __builtin_amdgcn_global_load_async_to_lds_b128(
                    (gv4i_p)(v4i*)const_cast<_Float16*>(vbase + s * kC + ch * 8),
                    (lv4i_p)(v4i*)(vrow + s * kKsStride + ch * 8), 0, 0);
#else
                *(v8h*)(ks + s * kKsStride + ch * 8)   = *(const v8h*)(kbase + s * kC + ch * 8);
                *(v8h*)(vrow + s * kKsStride + ch * 8) = *(const v8h*)(vbase + s * kC + ch * 8);
#endif
            }
        }

#if HAVE_ASYNC_LDS
#if __has_builtin(__builtin_amdgcn_s_wait_asynccnt)
        __builtin_amdgcn_s_wait_asynccnt(0);
#else
        asm volatile("s_wait_asynccnt 0" ::: "memory");
#endif
#endif
        __syncthreads();   // B: ks / vrow ready

        // ---- S = Q K^T : issue 16 WMMAs (B operands conflict-free in LDS) ----
        v8f sacc0 = zero8(), sacc1 = zero8();
        {
            const _Float16* k0 = ks + lid * kKsStride + bbase;
            const _Float16* k1 = ks + (16 + lid) * kKsStride + bbase;
#pragma unroll
            for (int kk = 0; kk < 8; ++kk) {
                sacc0 = WMMA16(qa[kk], load16c(k0 + kk * 32), sacc0);
                sacc1 = WMMA16(qa[kk], load16c(k1 + kk * 32), sacc1);
            }
        }

        // ---- transpose V: vrow -> vt, packed b32 stores (co-executes w/ XDL) ----
#pragma unroll
        for (int rep = 0; rep < 4; ++rep) {
            const int c  = rep * 128 + tid;      // 0..511
            const int sp = c & 15;               // s-pair
            const int ch = c >> 4;               // d-chunk of 8
            const v8h lo = *(const v8h*)(vrow + (2 * sp)     * kKsStride + ch * 8);
            const v8h hi = *(const v8h*)(vrow + (2 * sp + 1) * kKsStride + ch * 8);
#pragma unroll
            for (int j = 0; j < 8; ++j) {
                union { _Float16 h[2]; unsigned u; } pk;
                pk.h[0] = lo[j]; pk.h[1] = hi[j];
                *(unsigned*)(vt + (ch * 8 + j) * kVtStride + 2 * sp) = pk.u;
            }
        }

        // ---- scale + causal mask ----
#pragma unroll
        for (int r = 0; r < 8; ++r) {
            const int t = t0 + r + half_id * 8;
            float v0 = sacc0[r] * scale;
            float v1 = sacc1[r] * scale;
            if (s0 + lid      > t) v0 = -1e30f;
            if (s0 + 16 + lid > t) v1 = -1e30f;
            sacc0[r] = v0; sacc1[r] = v1;
        }

        // ---- online softmax ----
        float corrf[8];
#pragma unroll
        for (int r = 0; r < 8; ++r) {
            float rm = fmaxf(sacc0[r], sacc1[r]);
#pragma unroll
            for (int off = 1; off < 16; off <<= 1)
                rm = fmaxf(rm, __shfl_xor(rm, off, 32));
            const float mnew = fmaxf(mrow[r], rm);
            const float corr = __expf(mrow[r] - mnew);
            const float p0 = __expf(sacc0[r] - mnew);
            const float p1 = __expf(sacc1[r] - mnew);
            float ps = p0 + p1;
#pragma unroll
            for (int off = 1; off < 16; off <<= 1)
                ps += __shfl_xor(ps, off, 32);
            lrow[r]  = lrow[r] * corr + ps;
            mrow[r]  = mnew;
            corrf[r] = corr;
            sacc0[r] = p0; sacc1[r] = p1;
        }
#pragma unroll
        for (int n = 0; n < 16; ++n)
#pragma unroll
            for (int r = 0; r < 8; ++r) oacc[n][r] *= corrf[r];

        // ---- repack P (16x32) C-layout -> A-layout via wave-private LDS ----
#pragma unroll
        for (int r = 0; r < 8; ++r) {
            const int m = r + half_id * 8;
            pl[m * kPStride + lid]      = (_Float16)sacc0[r];
            pl[m * kPStride + 16 + lid] = (_Float16)sacc1[r];
        }
        v16h pa;
#pragma unroll
        for (int i = 0; i < 8; ++i) {
            pa[i]     = pl[lid * kPStride + bbase + i];
            pa[i + 8] = pl[lid * kPStride + bbase + 8 + i];
        }

        __syncthreads();   // C: vt writes visible to all waves

        // ---- O += P * V : B operand from V^T LDS (conflict-free) ----
#pragma unroll
        for (int n = 0; n < 16; ++n) {
            const _Float16* vp = vt + (n * 16 + lid) * kVtStride + bbase;
            oacc[n] = WMMA16(pa, load16c(vp), oacc[n]);
        }
    }

    // ---- normalize + store fp32 ----
    float linv[8];
#pragma unroll
    for (int r = 0; r < 8; ++r) linv[r] = 1.0f / lrow[r];
#pragma unroll
    for (int n = 0; n < 16; ++n)
#pragma unroll
        for (int r = 0; r < 8; ++r) {
            const int t = t0 + r + half_id * 8;
            out[(rowbase + t) * kC + n * 16 + lid] = oacc[n][r] * linv[r];
        }
}

// ---------------------------------------------------------------------------
// Launcher
// ---------------------------------------------------------------------------
extern "C" void kernel_launch(void* const* d_in, const int* in_sizes, int n_in,
                              void* d_out, int out_size, void* d_ws, size_t ws_size,
                              hipStream_t stream) {
    const float* x  = (const float*)d_in[0];
    const float* Wq = (const float*)d_in[1];
    const float* bq = (const float*)d_in[2];
    const float* Wk = (const float*)d_in[3];
    const float* bk = (const float*)d_in[4];
    const float* Wv = (const float*)d_in[5];
    const float* bv = (const float*)d_in[6];
    float* out = (float*)d_out;

    _Float16* qh = (_Float16*)d_ws;
    _Float16* kh = qh + (size_t)kRows * kC;
    _Float16* vh = kh + (size_t)kRows * kC;

    dim3 pgrid(kRows / 16 / 8, kC / 16, 3);
    qkv_proj_kernel<<<pgrid, 256, 0, stream>>>(x, Wq, bq, Wk, bk, Wv, bv, qh, kh, vh);

    dim3 agrid(kRows / 64, 1, 1);
    flash_attn_kernel<<<agrid, 128, 0, stream>>>(qh, kh, vh, out);
}
